// MoREnet_55336358642695
// MI455X (gfx1250) — matrix operand order repaired and checked by reading, not compiled, not run on hardware
//
#include <hip/hip_runtime.h>

// ---------------------------------------------------------------------------
// MoE (shared MLP + 4 dense experts, top-2 gating) for MI455X / gfx1250.
// All GEMMs on v_wmma_f32_16x16x32_bf16 (wave32 WMMA). Weights pre-transposed
// to bf16 [N][K] in d_ws so B-fragments are contiguous per-lane loads.
// M-tile = 32 tokens: each B-fragment feeds 2 WMMAs (2 independent acc
// chains per k-step -> hides the 5-slot bf16 WMMA RAW hazard, halves
// weight-fetch per FLOP).
// ---------------------------------------------------------------------------

typedef __bf16 bf16_t;
typedef __bf16 bf16x8  __attribute__((ext_vector_type(8)));
typedef __bf16 bf16x16 __attribute__((ext_vector_type(16)));
typedef float  f32x8   __attribute__((ext_vector_type(8)));

union FragU { bf16x16 v; bf16x8 h[2]; };

#define NTOK 32768   // B*A
#define MT   32      // tokens per workgroup tile
#define DIN  256
#define DOUT 256
#define DSH1 256
#define DSH2 128
#define DEH1 128
#define DEH2 64
#define NEXP 4

// d_ws layout (bf16 elements), all matrices stored transposed: [N][K]
#define OFF_SW1 0                           // [256][256]
#define OFF_SW2 (OFF_SW1 + DSH1 * DIN)      // [128][256]
#define OFF_SW3 (OFF_SW2 + DSH2 * DSH1)     // [256][128]
#define OFF_EW1 (OFF_SW3 + DOUT * DSH2)     // 4 x [128][256]
#define OFF_EW2 (OFF_EW1 + NEXP * DEH1 * DIN)   // 4 x [64][128]
#define OFF_EW3 (OFF_EW2 + NEXP * DEH2 * DEH1)  // 4 x [256][64]

__device__ __forceinline__ bf16_t f2bf(float f) {
  union { float f; unsigned u; } x; x.f = f;
  unsigned r = x.u + 0x7fffu + ((x.u >> 16) & 1u);   // round-to-nearest-even
  unsigned short hs = (unsigned short)(r >> 16);
  return __builtin_bit_cast(bf16_t, hs);
}
__device__ __forceinline__ float bf2f(bf16_t b) {
  unsigned short hs = __builtin_bit_cast(unsigned short, b);
  union { unsigned u; float f; } x; x.u = ((unsigned)hs) << 16;
  return x.f;
}

// Two 16-row M-tiles (rows 0-15 and 16-31 of A) x one 16-col N-tile of Bt.
// A row-major in LDS (ld=lda); Bt transposed weight [N][K] in global (L2-hot).
// B fragment is loaded once per k-step and feeds both WMMAs.
__device__ __forceinline__ void wmma_gemm2(const bf16_t* __restrict__ A, int lda,
                                           const bf16_t* __restrict__ Bt, int ldb,
                                           int n0, int ksteps, int lane,
                                           f32x8& acc0, f32x8& acc1) {
  const int m  = lane & 15;
  const int ka = (lane >> 4) << 3;   // A layout: K split 8/8 at +0/+16
  const int kb = (lane >> 4) << 3;   // B layout: K split 8/8 at +0/+16
  const bf16_t* ap0 = A + m * lda + ka;
  const bf16_t* ap1 = ap0 + 16 * lda;
  const bf16_t* bp  = Bt + (long)(n0 + m) * ldb + kb;
  for (int ks = 0; ks < ksteps; ++ks) {
    FragU a0, a1, b;
    b.h[0]  = *(const bf16x8*)(bp);
    b.h[1]  = *(const bf16x8*)(bp + 16);
    a0.h[0] = *(const bf16x8*)(ap0);
    a0.h[1] = *(const bf16x8*)(ap0 + 16);
    a1.h[0] = *(const bf16x8*)(ap1);
    a1.h[1] = *(const bf16x8*)(ap1 + 16);
    acc0 = __builtin_amdgcn_wmma_f32_16x16x32_bf16(false, a0.v, false, b.v,
                                                   (short)0, acc0, false, false);
    acc1 = __builtin_amdgcn_wmma_f32_16x16x32_bf16(false, a1.v, false, b.v,
                                                   (short)0, acc1, false, false);
    ap0 += 32; ap1 += 32; bp += 32;
  }
}

// relu(acc + bias) -> bf16 activation buffer, both M-tiles.
__device__ __forceinline__ void store_act(const f32x8& a0, const f32x8& a1,
                                          bf16_t* H, int ldh, int n, int mb,
                                          float bn) {
  for (int r = 0; r < 8; ++r) {
    float v0 = a0[r] + bn; v0 = v0 > 0.f ? v0 : 0.f;
    float v1 = a1[r] + bn; v1 = v1 > 0.f ? v1 : 0.f;
    H[(mb + r) * ldh + n]      = f2bf(v0);
    H[(16 + mb + r) * ldh + n] = f2bf(v1);
  }
}

// f32 [K][N] (optionally batched) -> bf16 [N][K]
__global__ void transpose_to_bf16(const float* __restrict__ src,
                                  bf16_t* __restrict__ dst, int K, int N) {
  long base = (long)blockIdx.y * K * N;
  int i = blockIdx.x * blockDim.x + threadIdx.x;
  if (i < K * N) {
    int k = i / N, n = i - k * N;
    dst[base + (long)n * K + k] = f2bf(src[base + i]);
  }
}

__global__ __launch_bounds__(256)
void moe_fused_kernel(const float* __restrict__ x,
                      const float* __restrict__ sb1, const float* __restrict__ sb2,
                      const float* __restrict__ sb3,
                      const float* __restrict__ eb1, const float* __restrict__ eb2,
                      const float* __restrict__ eb3,
                      const float* __restrict__ gW,  const float* __restrict__ gb,
                      const bf16_t* __restrict__ wt,
                      float* __restrict__ out) {
  __shared__ __align__(16) bf16_t xb[MT * DIN];    // 16 KB  x tile (bf16)
  __shared__ __align__(16) bf16_t hA[MT * DSH1];   // 16 KB  shared-h1 / expert-h1
  __shared__ __align__(16) bf16_t hB[MT * DSH2];   //  8 KB  shared-h2 / expert-h2
  __shared__ __align__(16) float  outf[MT * DOUT]; // 32 KB  f32 output accumulator
  __shared__ float glog[MT * NEXP];
  __shared__ float wgt[MT * NEXP];

  const int tid  = threadIdx.x;
  const int w    = tid >> 5;
  const int lane = tid & 31;
  const int m15  = lane & 15;
  const int mb   = (lane >> 4) * 8;
  const long t0  = (long)blockIdx.x * MT;

  // ---- Stage 0: load x tile -> bf16 LDS (coalesced) ----
  const float* xrow = x + t0 * DIN;
  for (int i = tid; i < MT * DIN; i += 256) xb[i] = f2bf(xrow[i]);
  __syncthreads();

  // ---- Gate logits: 128 (token,expert) dots on waves 0-3 ----
  if (tid < MT * NEXP) {
    int t = tid >> 2, e = tid & 3;
    float g = gb[e];
    for (int k = 0; k < DIN; ++k)
      g += bf2f(xb[t * DIN + k]) * gW[k * NEXP + e];
    glog[t * NEXP + e] = g;
  }
  __syncthreads();

  // ---- Gate softmax + top-2 renormalization (1 lane / token) ----
  if (tid < MT) {
    int t = tid;
    float p[4];
    float g0 = glog[t*4+0], g1 = glog[t*4+1], g2 = glog[t*4+2], g3 = glog[t*4+3];
    float mx = fmaxf(fmaxf(g0, g1), fmaxf(g2, g3));
    p[0] = __expf(g0 - mx); p[1] = __expf(g1 - mx);
    p[2] = __expf(g2 - mx); p[3] = __expf(g3 - mx);
    float inv = 1.0f / (p[0] + p[1] + p[2] + p[3]);
    p[0] *= inv; p[1] *= inv; p[2] *= inv; p[3] *= inv;
    int i1 = 0;
    for (int e = 1; e < 4; ++e) if (p[e] > p[i1]) i1 = e;
    int i2 = (i1 == 0) ? 1 : 0;
    for (int e = 0; e < 4; ++e) if (e != i1 && p[e] > p[i2]) i2 = e;
    float ex1 = __expf(p[i1]), ex2 = __expf(p[i2]);
    float rn = 1.0f / (ex1 + ex2);
    wgt[t*4+0] = 0.f; wgt[t*4+1] = 0.f; wgt[t*4+2] = 0.f; wgt[t*4+3] = 0.f;
    wgt[t*4+i1] = ex1 * rn;
    wgt[t*4+i2] = ex2 * rn;
  }
  __syncthreads();

  // ---- Shared L1: hA[32x256] = relu(xb @ sW1 + sb1); 2 N-tiles / wave ----
  for (int tt = 0; tt < 2; ++tt) {
    int n0 = (w * 2 + tt) * 16;
    f32x8 a0 = {0,0,0,0,0,0,0,0}, a1 = {0,0,0,0,0,0,0,0};
    wmma_gemm2(xb, DIN, wt + OFF_SW1, DIN, n0, DIN / 32, lane, a0, a1);
    store_act(a0, a1, hA, DSH1, n0 + m15, mb, sb1[n0 + m15]);
  }
  __syncthreads();

  // ---- Shared L2: hB[32x128] = relu(hA @ sW2 + sb2); 1 N-tile / wave ----
  {
    int n0 = w * 16;
    f32x8 a0 = {0,0,0,0,0,0,0,0}, a1 = {0,0,0,0,0,0,0,0};
    wmma_gemm2(hA, DSH1, wt + OFF_SW2, DSH1, n0, DSH1 / 32, lane, a0, a1);
    store_act(a0, a1, hB, DSH2, n0 + m15, mb, sb2[n0 + m15]);
  }
  __syncthreads();

  // ---- Shared L3: outf = hB @ sW3 + sb3 (f32, no relu) ----
  for (int tt = 0; tt < 2; ++tt) {
    int n0 = (w * 2 + tt) * 16;
    f32x8 a0 = {0,0,0,0,0,0,0,0}, a1 = {0,0,0,0,0,0,0,0};
    wmma_gemm2(hB, DSH2, wt + OFF_SW3, DSH2, n0, DSH2 / 32, lane, a0, a1);
    int n = n0 + m15;
    float bn = sb3[n];
    for (int r = 0; r < 8; ++r) {
      outf[(mb + r) * DOUT + n]      = a0[r] + bn;
      outf[(16 + mb + r) * DOUT + n] = a1[r] + bn;
    }
  }
  __syncthreads();

  // ---- Experts: outf += wgt[:,e] * expert_e(x) ----
  for (int e = 0; e < NEXP; ++e) {
    {   // L1: hA[32x128] = relu(xb @ eW1[e] + eb1[e]); 1 tile / wave
      int n0 = w * 16;
      f32x8 a0 = {0,0,0,0,0,0,0,0}, a1 = {0,0,0,0,0,0,0,0};
      wmma_gemm2(xb, DIN, wt + OFF_EW1 + e * (DEH1 * DIN), DIN,
                 n0, DIN / 32, lane, a0, a1);
      store_act(a0, a1, hA, DEH1, n0 + m15, mb, eb1[e * DEH1 + n0 + m15]);
    }
    __syncthreads();
    if (w < 4) {   // L2: hB[32x64]; waves 0-3 (wave-uniform branch, EXEC all-1)
      int n0 = w * 16;
      f32x8 a0 = {0,0,0,0,0,0,0,0}, a1 = {0,0,0,0,0,0,0,0};
      wmma_gemm2(hA, DEH1, wt + OFF_EW2 + e * (DEH2 * DEH1), DEH1,
                 n0, DEH1 / 32, lane, a0, a1);
      store_act(a0, a1, hB, DEH2, n0 + m15, mb, eb2[e * DEH2 + n0 + m15]);
    }
    __syncthreads();
    for (int tt = 0; tt < 2; ++tt) {   // L3: gated accumulate into outf
      int n0 = (w * 2 + tt) * 16;
      f32x8 a0 = {0,0,0,0,0,0,0,0}, a1 = {0,0,0,0,0,0,0,0};
      wmma_gemm2(hB, DEH2, wt + OFF_EW3 + e * (DOUT * DEH2), DEH2,
                 n0, DEH2 / 32, lane, a0, a1);
      int n = n0 + m15;
      float bn = eb3[e * DOUT + n];
      for (int r = 0; r < 8; ++r) {
        float w0 = wgt[(mb + r) * 4 + e];
        float w1 = wgt[(16 + mb + r) * 4 + e];
        outf[(mb + r) * DOUT + n]      += w0 * (a0[r] + bn);
        outf[(16 + mb + r) * DOUT + n] += w1 * (a1[r] + bn);
      }
    }
    __syncthreads();
  }

  // ---- Store tile (coalesced f32) ----
  float* orow = out + t0 * DOUT;
  for (int i = tid; i < MT * DOUT; i += 256) orow[i] = outf[i];
}

extern "C" void kernel_launch(void* const* d_in, const int* in_sizes, int n_in,
                              void* d_out, int out_size, void* d_ws, size_t ws_size,
                              hipStream_t stream) {
  (void)in_sizes; (void)n_in; (void)out_size; (void)ws_size;
  const float* x   = (const float*)d_in[0];
  const float* sW1 = (const float*)d_in[1];
  const float* sb1 = (const float*)d_in[2];
  const float* sW2 = (const float*)d_in[3];
  const float* sb2 = (const float*)d_in[4];
  const float* sW3 = (const float*)d_in[5];
  const float* sb3 = (const float*)d_in[6];
  const float* eW1 = (const float*)d_in[7];
  const float* eb1 = (const float*)d_in[8];
  const float* eW2 = (const float*)d_in[9];
  const float* eb2 = (const float*)d_in[10];
  const float* eW3 = (const float*)d_in[11];
  const float* eb3 = (const float*)d_in[12];
  const float* gW  = (const float*)d_in[13];
  const float* gb  = (const float*)d_in[14];
  float*  out = (float*)d_out;
  bf16_t* wt  = (bf16_t*)d_ws;

  dim3 blk(256);
  // Pre-transpose + bf16-convert all weights into d_ws (L2-resident, ~0.7 MB).
  transpose_to_bf16<<<dim3((DIN  * DSH1 + 255) / 256, 1),    blk, 0, stream>>>(sW1, wt + OFF_SW1, DIN,  DSH1);
  transpose_to_bf16<<<dim3((DSH1 * DSH2 + 255) / 256, 1),    blk, 0, stream>>>(sW2, wt + OFF_SW2, DSH1, DSH2);
  transpose_to_bf16<<<dim3((DSH2 * DOUT + 255) / 256, 1),    blk, 0, stream>>>(sW3, wt + OFF_SW3, DSH2, DOUT);
  transpose_to_bf16<<<dim3((DIN  * DEH1 + 255) / 256, NEXP), blk, 0, stream>>>(eW1, wt + OFF_EW1, DIN,  DEH1);
  transpose_to_bf16<<<dim3((DEH1 * DEH2 + 255) / 256, NEXP), blk, 0, stream>>>(eW2, wt + OFF_EW2, DEH1, DEH2);
  transpose_to_bf16<<<dim3((DEH2 * DOUT + 255) / 256, NEXP), blk, 0, stream>>>(eW3, wt + OFF_EW3, DEH2, DOUT);

  moe_fused_kernel<<<dim3(NTOK / MT), blk, 0, stream>>>(
      x, sb1, sb2, sb3, eb1, eb2, eb3, gW, gb, wt, out);
}
